// MinCostFlowModel_7378753815339
// MI455X (gfx1250) — matrix operand order repaired and checked by reading, not compiled, not run on hardware
//
#include <hip/hip_runtime.h>
#include <hip/hip_bf16.h>
#include <math.h>

// Problem constants (match reference)
#define BQ 8
#define NQ 1024
#define DQ 16
#define EQ 64
#define HQ 4
#define LEAKQ 0.2f
#define BIGQ 1.0e7f
#define THRQ 1.0e-4f
#define ITERSQ 20

typedef __attribute__((ext_vector_type(16))) __bf16 v16bf;
typedef __attribute__((ext_vector_type(8)))  float v8f;

__device__ __forceinline__ unsigned short f2bf_bits(float f) {
  union { float f; unsigned u; } v; v.f = f;
  unsigned r = v.u + 0x7fffu + ((v.u >> 16) & 1u);   // RNE
  return (unsigned short)(r >> 16);
}
__device__ __forceinline__ __bf16 bits2bf(unsigned short s) {
  union { unsigned short u; __bf16 h; } v; v.u = s; return v.h;
}
__device__ __forceinline__ v8f wmma_bf16(v16bf a, v16bf b, v8f c) {
  return __builtin_amdgcn_wmma_f32_16x16x32_bf16(false, a, false, b, (short)0, c, false, false);
}
// A-fragment K index for 16-bit 16x32 A: lanes 0-15 hold K {0..7,16..23}, lanes 16-31 +8.
__device__ __forceinline__ int a_kidx(int e, int kh) { return ((e < 8) ? e : (e + 8)) + kh * 8; }

// ---------------------------------------------------------------- weights -> bf16
__global__ void k_convert(const float* __restrict__ gw, const float* __restrict__ dw,
                          const float* __restrict__ bw, unsigned short* gwB,
                          unsigned short* dwB, unsigned short* bwB) {
  int i = blockIdx.x * 256 + threadIdx.x;
  if (i < HQ * EQ * EQ) gwB[i] = f2bf_bits(gw[i]);
  if (i < EQ * NQ) { dwB[i] = f2bf_bits(dw[i]); bwB[i] = f2bf_bits(bw[i]); }
}

// ---------------------------------------------------------------- encoder (K=17, tiny)
__global__ void k_encoder(const float* __restrict__ xin, const float* __restrict__ emb,
                          const float* __restrict__ W, const float* __restrict__ bvec,
                          float* __restrict__ enc, unsigned short* __restrict__ encB) {
  int idx = blockIdx.x * 256 + threadIdx.x;  // b*N + n
  if (idx >= BQ * NQ) return;
  int n = idx % NQ;
  float in[1 + DQ];
  in[0] = xin[idx];
  for (int d = 0; d < DQ; ++d) in[1 + d] = emb[n * DQ + d];
  for (int e = 0; e < EQ; ++e) {
    float s = bvec[e];
    for (int k = 0; k < 1 + DQ; ++k) s += in[k] * W[k * EQ + e];
    s = s > 0.f ? s : 0.f;
    enc[(size_t)idx * EQ + e] = s;
    encB[(size_t)idx * EQ + e] = f2bf_bits(s);
  }
}

// ---------------------------------------------------------------- h = enc @ W_h + b, plus f_s/f_n
__global__ void k_gat_h(const unsigned short* __restrict__ encB,
                        const unsigned short* __restrict__ WB,
                        const float* __restrict__ gb, const float* __restrict__ aS,
                        const float* __restrict__ aN, unsigned short* __restrict__ hB,
                        float* __restrict__ fS, float* __restrict__ fN) {
  int wave = threadIdx.x >> 5, lane = threadIdx.x & 31;
  int tile = blockIdx.x & 7, bh = blockIdx.x >> 3;
  int hh = bh % HQ, b = bh / HQ;
  int mBase = tile * 128 + wave * 16;
  int kh = lane >> 4, l15 = lane & 15;
  int rowA = mBase + l15;
  v8f z = {0.f, 0.f, 0.f, 0.f, 0.f, 0.f, 0.f, 0.f};
  v8f c[4] = {z, z, z, z};
  for (int kt = 0; kt < 2; ++kt) {
    int kBase = kt * 32;
    v16bf a;
    const unsigned short* ap = encB + ((size_t)(b * NQ + rowA)) * EQ + kBase + kh * 8;
#pragma unroll
    for (int e = 0; e < 16; ++e) a[e] = bits2bf(ap[(e < 8) ? e : (e + 8)]);
    int krow = kBase;
    const unsigned short* wp = WB + ((size_t)hh * EQ + krow) * EQ + l15;
#pragma unroll
    for (int t = 0; t < 4; ++t) {
      v16bf bf;
#pragma unroll
      for (int e = 0; e < 16; ++e) bf[e] = bits2bf(wp[a_kidx(e, kh) * EQ + t * 16]);
      c[t] = wmma_bf16(a, bf, c[t]);
    }
  }
  float fsAcc[8], fnAcc[8];
#pragma unroll
  for (int r = 0; r < 8; ++r) { fsAcc[r] = 0.f; fnAcc[r] = 0.f; }
#pragma unroll
  for (int t = 0; t < 4; ++t) {
    int n = t * 16 + l15;
    float bb = gb[hh * EQ + n], as = aS[hh * EQ + n], an = aN[hh * EQ + n];
#pragma unroll
    for (int r = 0; r < 8; ++r) {
      float v = c[t][r] + bb;
      c[t][r] = v;
      fsAcc[r] += v * as;
      fnAcc[r] += v * an;
    }
  }
  size_t bhN = (size_t)(b * HQ + hh) * NQ;
#pragma unroll
  for (int r = 0; r < 8; ++r) {
    float vs = fsAcc[r], vn = fnAcc[r];
    for (int m = 1; m <= 8; m <<= 1) { vs += __shfl_xor(vs, m, 32); vn += __shfl_xor(vn, m, 32); }
    if (l15 == 0) {
      int m = mBase + r + 8 * kh;
      fS[bhN + m] = vs;
      fN[bhN + m] = vn;
    }
  }
#pragma unroll
  for (int t = 0; t < 4; ++t)
#pragma unroll
    for (int r = 0; r < 8; ++r) {
      int m = mBase + r + 8 * kh;
      hB[(bhN + m) * EQ + t * 16 + l15] = f2bf_bits(c[t][r]);
    }
}

// ---------------------------------------------------------------- flash-style GAT attention
// One block = (b, 16-row tile); one wave per head. The head-invariant bias tile
// is loaded coalesced into LDS once and shared by all 4 heads (4x traffic cut).
__global__ void k_gat_attn(const unsigned short* __restrict__ hB,
                           const float* __restrict__ fS, const float* __restrict__ fN,
                           const float* __restrict__ nbias, float* __restrict__ attnO) {
  __shared__ float bias_t[16 * 33];  // 16 rows x 32 j, stride-33 pad (bank-conflict free)
  int wave = threadIdx.x >> 5, lane = threadIdx.x & 31;
  int b = blockIdx.x >> 6, mTile = blockIdx.x & 63;
  int hh = wave;  // head per wave
  int mBase = mTile * 16;
  int kh = lane >> 4, l15 = lane & 15;
  int mA = mBase + l15;
  size_t bhN = (size_t)(b * HQ + hh) * NQ;
  float fsv = fS[bhN + mA];
  float rowMax = -3.0e38f, denom = 0.f;
  v8f z = {0.f, 0.f, 0.f, 0.f, 0.f, 0.f, 0.f, 0.f};
  v8f acc[4] = {z, z, z, z};
  for (int jt = 0; jt < NQ / 32; ++jt) {
    int jBase = jt * 32;
    // cooperative coalesced load of the 16x32 bias tile (512 floats / 128 threads)
#pragma unroll
    for (int k = 0; k < 4; ++k) {
      int idx = threadIdx.x + k * 128;
      int rr = idx >> 5, jj = idx & 31;
      bias_t[rr * 33 + jj] = nbias[((size_t)b * NQ + mBase + rr) * NQ + jBase + jj];
    }
    __syncthreads();
    float l[16];
    float tmax = -3.0e38f;
#pragma unroll
    for (int e = 0; e < 16; ++e) {
      int jl = a_kidx(e, kh);
      float s = fsv + fN[bhN + jBase + jl];
      s = s > 0.f ? s : LEAKQ * s;
      s += bias_t[l15 * 33 + jl];
      l[e] = s;
      tmax = fmaxf(tmax, s);
    }
    tmax = fmaxf(tmax, __shfl_xor(tmax, 16, 32));
    float newMax = fmaxf(rowMax, tmax);
    float corr = __expf(rowMax - newMax);
    v16bf p;
    float tsum = 0.f;
#pragma unroll
    for (int e = 0; e < 16; ++e) {
      float pv = __expf(l[e] - newMax);
      tsum += pv;
      p[e] = (__bf16)pv;
    }
    tsum += __shfl_xor(tsum, 16, 32);
    denom = denom * corr + tsum;
    rowMax = newMax;
#pragma unroll
    for (int r = 0; r < 8; ++r) {
      float cr = __shfl(corr, r + 8 * kh, 32);  // per-C-row online-softmax rescale
      acc[0][r] *= cr; acc[1][r] *= cr; acc[2][r] *= cr; acc[3][r] *= cr;
    }
    const unsigned short* hp = hB + (bhN + jBase) * EQ + l15;
#pragma unroll
    for (int t = 0; t < 4; ++t) {
      v16bf bf;
#pragma unroll
      for (int e = 0; e < 16; ++e) bf[e] = bits2bf(hp[a_kidx(e, kh) * EQ + t * 16]);
      acc[t] = wmma_bf16(p, bf, acc[t]);
    }
    __syncthreads();  // protect bias_t before next iteration's overwrite
  }
#pragma unroll
  for (int r = 0; r < 8; ++r) {
    float inv = 1.f / __shfl(denom, r + 8 * kh, 32);
    int m = mBase + r + 8 * kh;
#pragma unroll
    for (int t = 0; t < 4; ++t) attnO[(bhN + m) * EQ + t * 16 + l15] = acc[t][r] * inv;
  }
}

// ---------------------------------------------------------------- head-mean + relu + gate
__global__ void k_gate(const float* __restrict__ attnO, float* __restrict__ enc,
                       unsigned short* __restrict__ encB, const float* __restrict__ w1,
                       const float* __restrict__ w2, const float* __restrict__ gb) {
  size_t idx = (size_t)blockIdx.x * 256 + threadIdx.x;  // over B*N*E
  int e = idx % EQ;
  size_t bn = idx / EQ;
  int b = bn / NQ, n = bn % NQ;
  float cur = 0.f;
  for (int h = 0; h < HQ; ++h) cur += attnO[(((size_t)b * HQ + h) * NQ + n) * EQ + e];
  cur *= (1.f / HQ);
  cur = cur > 0.f ? cur : 0.f;
  float prev = enc[idx];
  float zg = 1.f / (1.f + __expf(-(prev * w1[e] + cur * w2[e] + gb[e])));
  float o = zg * prev + (1.f - zg) * cur;
  enc[idx] = o;
  encB[idx] = f2bf_bits(o);
}

// ---------------------------------------------------------------- pred GEMM + row softmax (fused)
__global__ void k_pred_softmax(const unsigned short* __restrict__ encB,
                               const unsigned short* __restrict__ dWB,
                               const float* __restrict__ db, const float* __restrict__ nbias,
                               float* __restrict__ fwp) {
  __shared__ float lds[16 * NQ];  // 64 KB row block
  int b = blockIdx.x >> 6, mBase = (blockIdx.x & 63) * 16;
  int wave = threadIdx.x >> 5, lane = threadIdx.x & 31;
  int kh = lane >> 4, l15 = lane & 15;
  int nBase = wave * 128;
  v8f z = {0.f, 0.f, 0.f, 0.f, 0.f, 0.f, 0.f, 0.f};
  v8f c[8] = {z, z, z, z, z, z, z, z};
  for (int kt = 0; kt < 2; ++kt) {
    int kBase = kt * 32;
    v16bf a;
    const unsigned short* ap = encB + ((size_t)(b * NQ + mBase + l15)) * EQ + kBase + kh * 8;
#pragma unroll
    for (int e = 0; e < 16; ++e) a[e] = bits2bf(ap[(e < 8) ? e : (e + 8)]);
    int krow = kBase;
    const unsigned short* wp = dWB + (size_t)krow * NQ + nBase + l15;
#pragma unroll
    for (int t = 0; t < 8; ++t) {
      v16bf bf;
#pragma unroll
      for (int e = 0; e < 16; ++e) bf[e] = bits2bf(wp[(size_t)a_kidx(e, kh) * NQ + t * 16]);
      c[t] = wmma_bf16(a, bf, c[t]);
    }
  }
#pragma unroll
  for (int t = 0; t < 8; ++t)
#pragma unroll
    for (int r = 0; r < 8; ++r) {
      int m = mBase + r + 8 * kh;
      int n = nBase + t * 16 + l15;
      float v = c[t][r] + db[n] + nbias[((size_t)b * NQ + m) * NQ + n];
      if (n == m) v -= BIGQ;
      lds[(r + 8 * kh) * NQ + n] = v;
    }
  __syncthreads();
  int row = threadIdx.x >> 4, sub = threadIdx.x & 15;  // 16 threads per row
  float mx = -3.0e38f;
  for (int j = sub; j < NQ; j += 16) mx = fmaxf(mx, lds[row * NQ + j]);
  for (int m = 1; m <= 8; m <<= 1) mx = fmaxf(mx, __shfl_xor(mx, m, 32));
  float sm = 0.f;
  for (int j = sub; j < NQ; j += 16) {
    float ev = __expf(lds[row * NQ + j] - mx);
    lds[row * NQ + j] = ev;
    sm += ev;
  }
  for (int m = 1; m <= 8; m <<= 1) sm += __shfl_xor(sm, m, 32);
  float inv = 1.f / sm;
  for (int j = sub; j < NQ; j += 16)
    fwp[((size_t)b * NQ + mBase + row) * NQ + j] = lds[row * NQ + j] * inv;
}

// ---------------------------------------------------------------- alphas = enc @ alpha_W + b
__global__ void k_alphas(const float* __restrict__ enc, const float* __restrict__ aW,
                         const float* __restrict__ ab, float* __restrict__ alphas) {
  int idx = blockIdx.x * 256 + threadIdx.x;
  if (idx >= BQ * NQ) return;
  float s = ab[0];
  const float* ep = enc + (size_t)idx * EQ;
  for (int e = 0; e < EQ; ++e) s += ep[e] * aW[e];
  alphas[idx] = s;
}

// ---------------------------------------------------------------- state init (every call; ws is poisoned)
__global__ void k_init(float* flow0, float* dualIn, float* dualOut, float* scal,
                       int* flags, float* out) {
  size_t idx = (size_t)blockIdx.x * 256 + threadIdx.x;
  if (idx < (size_t)BQ * NQ * NQ) flow0[idx] = 0.f;
  if (idx < BQ * NQ) { dualIn[idx] = 0.f; dualOut[idx] = 0.f; }
  if (idx < 2 * BQ) scal[idx] = 0.f;
  if (idx < 32) flags[idx] = (idx == 0) ? 1 : 0;
  if (idx < BQ) out[(size_t)BQ * NQ * NQ + idx] = 0.f;  // flow_cost accumulators
}

// ---------------------------------------------------------------- fused betas GEMM + dual reductions
__global__ void k_dual(const unsigned short* __restrict__ encB,
                       const unsigned short* __restrict__ bWB, const float* __restrict__ bb,
                       const float* __restrict__ adj, const float* __restrict__ alphas,
                       float* __restrict__ dualIn, float* __restrict__ dualOut,
                       float* __restrict__ scal) {
  __shared__ float lds[16 * NQ];
  int b = blockIdx.x >> 6, mBase = (blockIdx.x & 63) * 16;
  int wave = threadIdx.x >> 5, lane = threadIdx.x & 31;
  int kh = lane >> 4, l15 = lane & 15;
  int nBase = wave * 128;
  v8f z = {0.f, 0.f, 0.f, 0.f, 0.f, 0.f, 0.f, 0.f};
  v8f c[8] = {z, z, z, z, z, z, z, z};
  for (int kt = 0; kt < 2; ++kt) {
    int kBase = kt * 32;
    v16bf a;
    const unsigned short* ap = encB + ((size_t)(b * NQ + mBase + l15)) * EQ + kBase + kh * 8;
#pragma unroll
    for (int e = 0; e < 16; ++e) a[e] = bits2bf(ap[(e < 8) ? e : (e + 8)]);
    int krow = kBase;
    const unsigned short* wp = bWB + (size_t)krow * NQ + nBase + l15;
#pragma unroll
    for (int t = 0; t < 8; ++t) {
      v16bf bf;
#pragma unroll
      for (int e = 0; e < 16; ++e) bf[e] = bits2bf(wp[(size_t)a_kidx(e, kh) * NQ + t * 16]);
      c[t] = wmma_bf16(a, bf, c[t]);
    }
  }
#pragma unroll
  for (int t = 0; t < 8; ++t)
#pragma unroll
    for (int r = 0; r < 8; ++r) {
      int n = nBase + t * 16 + l15;
      float v = c[t][r] + bb[n];
      lds[(r + 8 * kh) * NQ + n] = v > 0.f ? v : 0.f;  // betas tile, never hits HBM
    }
  __syncthreads();
  int tid = threadIdx.x;
  float am[16];
#pragma unroll
  for (int i = 0; i < 16; ++i) am[i] = alphas[b * NQ + mBase + i];
  float rowp[16];
#pragma unroll
  for (int i = 0; i < 16; ++i) rowp[i] = 0.f;
  float ssq = 0.f, sbdf = 0.f;
  for (int j = tid; j < NQ; j += 256) {
    float aj = alphas[b * NQ + j];
    float csum = 0.f;
#pragma unroll
    for (int i = 0; i < 16; ++i) {
      float beta = lds[i * NQ + j];
      float df = 0.5f * (aj - am[i] + beta);
      df = df > 0.f ? df : 0.f;
      df *= adj[((size_t)b * NQ + mBase + i) * NQ + j];
      csum += df;
      rowp[i] += df;
      ssq += df * df;
      sbdf += beta * df;
    }
    atomicAdd(&dualIn[b * NQ + j], csum);
  }
  int lane2 = threadIdx.x & 31;
#pragma unroll
  for (int i = 0; i < 16; ++i) {
    float v = rowp[i];
    for (int m = 16; m >= 1; m >>= 1) v += __shfl_xor(v, m, 32);
    if (lane2 == 0) atomicAdd(&dualOut[b * NQ + mBase + i], v);
  }
  for (int m = 16; m >= 1; m >>= 1) { ssq += __shfl_xor(ssq, m, 32); sbdf += __shfl_xor(sbdf, m, 32); }
  if (lane2 == 0) { atomicAdd(&scal[b], ssq); atomicAdd(&scal[BQ + b], sbdf); }
}

__global__ void k_dual_final(const float* __restrict__ alphas, const float* __restrict__ dualIn,
                             const float* __restrict__ dualOut, const float* __restrict__ xin,
                             const float* __restrict__ scal, float* __restrict__ out) {
  int b = blockIdx.x, t = threadIdx.x;
  float sA = 0.f, sIn = 0.f, sAX = 0.f;
  for (int n = t; n < NQ; n += 256) {
    float a = alphas[b * NQ + n];
    sA += a;
    sIn += dualIn[b * NQ + n];
    sAX += a * (dualOut[b * NQ + n] + xin[b * NQ + n]);
  }
  __shared__ float r0[256], r1[256], r2[256];
  r0[t] = sA; r1[t] = sIn; r2[t] = sAX;
  __syncthreads();
  for (int s = 128; s > 0; s >>= 1) {
    if (t < s) { r0[t] += r0[t + s]; r1[t] += r1[t + s]; r2[t] += r2[t + s]; }
    __syncthreads();
  }
  if (t == 0) {
    float pen = r0[0] * r1[0] - (float)NQ * r2[0] - scal[BQ + b];
    out[(size_t)BQ * NQ * NQ + BQ + b] = scal[b] + pen;  // dual_cost
  }
}

// ---------------------------------------------------------------- flow fixed-point loop
__global__ void k_inflow(const float* __restrict__ adj, const float* __restrict__ cur,
                         const float* __restrict__ xin, float* __restrict__ rvec,
                         const int* __restrict__ flags, int it) {
  if (!flags[it]) return;
  int b = blockIdx.x >> 2;
  int j = ((blockIdx.x & 3) << 8) + threadIdx.x;
  const float* ap = adj + (size_t)b * NQ * NQ + j;
  const float* fp = cur + (size_t)b * NQ * NQ + j;
  float s = 0.f;
  for (int i = 0; i < NQ; ++i) s += ap[(size_t)i * NQ] * fp[(size_t)i * NQ];
  float r = s - xin[b * NQ + j];
  rvec[b * NQ + j] = r > 0.f ? r : 0.f;
}

__global__ void k_flow_update(const float* __restrict__ fwp, const float* __restrict__ rvec,
                              const float* __restrict__ cur, float* __restrict__ nxt,
                              int* __restrict__ flags, int it) {
  if (!flags[it]) return;
  size_t idx = (size_t)blockIdx.x * 256 + threadIdx.x;
  int b = idx / ((size_t)NQ * NQ);
  int i = (idx % ((size_t)NQ * NQ)) / NQ;
  float nf = fwp[idx] * rvec[b * NQ + i];
  float d = fabsf(nf - cur[idx]);
  nxt[idx] = nf;
  if (d > THRQ) flags[it + 1] = 1;  // benign racy all-write-1
}

__global__ void k_finalize(const float* __restrict__ buf0, const float* __restrict__ buf1,
                           const int* __restrict__ flags, float* __restrict__ out) {
  int k = 0;
  for (int t = 0; t < ITERSQ; ++t) k += flags[t];
  const float* src = (k & 1) ? buf1 : buf0;
  size_t idx = (size_t)blockIdx.x * 256 + threadIdx.x;
  float v = src[idx];
  out[idx] = v;  // flow output
  int b = idx / ((size_t)NQ * NQ);
  float sq = v * v;
  for (int m = 16; m >= 1; m >>= 1) sq += __shfl_xor(sq, m, 32);
  if ((threadIdx.x & 31) == 0) atomicAdd(&out[(size_t)BQ * NQ * NQ + b], sq);
}

__global__ void k_loss(float* __restrict__ out) {
  float acc = 0.f;
  for (int b = 0; b < BQ; ++b) {
    float fc = out[(size_t)BQ * NQ * NQ + b];
    float dc = out[(size_t)BQ * NQ * NQ + BQ + b];
    float d = fc - dc;
    acc += d * d + 0.1f * fc;
  }
  out[(size_t)BQ * NQ * NQ + 2 * BQ] = acc / (float)BQ;
}

// ---------------------------------------------------------------- host driver
extern "C" void kernel_launch(void* const* d_in, const int* in_sizes, int n_in,
                              void* d_out, int out_size, void* d_ws, size_t ws_size,
                              hipStream_t stream) {
  (void)in_sizes; (void)n_in; (void)out_size; (void)ws_size;
  const float* node_input = (const float*)d_in[0];
  const float* node_emb   = (const float*)d_in[1];
  const float* node_bias  = (const float*)d_in[2];
  const float* adj        = (const float*)d_in[3];
  const float* enc_W      = (const float*)d_in[4];
  const float* enc_b      = (const float*)d_in[5];
  const float* gat_W      = (const float*)d_in[6];
  const float* gat_b      = (const float*)d_in[7];
  const float* a_self     = (const float*)d_in[8];
  const float* a_neigh    = (const float*)d_in[9];
  const float* gate_w1    = (const float*)d_in[10];
  const float* gate_w2    = (const float*)d_in[11];
  const float* gate_b     = (const float*)d_in[12];
  const float* dec_W      = (const float*)d_in[13];
  const float* dec_b      = (const float*)d_in[14];
  const float* alpha_W    = (const float*)d_in[15];
  const float* alpha_b    = (const float*)d_in[16];
  const float* beta_W     = (const float*)d_in[17];
  const float* beta_b     = (const float*)d_in[18];
  float* out = (float*)d_out;

  char* w = (char*)d_ws;
  auto carve = [&](size_t bytes) {
    char* p = w;
    w += (bytes + 255) & ~(size_t)255;
    return p;
  };
  float*          encF   = (float*)carve((size_t)BQ * NQ * EQ * 4);
  unsigned short* encB   = (unsigned short*)carve((size_t)BQ * NQ * EQ * 2);
  unsigned short* gatWB  = (unsigned short*)carve((size_t)HQ * EQ * EQ * 2);
  unsigned short* decWB  = (unsigned short*)carve((size_t)EQ * NQ * 2);
  unsigned short* betWB  = (unsigned short*)carve((size_t)EQ * NQ * 2);
  unsigned short* hB     = (unsigned short*)carve((size_t)BQ * HQ * NQ * EQ * 2);
  float*          fS     = (float*)carve((size_t)BQ * HQ * NQ * 4);
  float*          fN     = (float*)carve((size_t)BQ * HQ * NQ * 4);
  float*          attnO  = (float*)carve((size_t)BQ * HQ * NQ * EQ * 4);
  float*          fwp    = (float*)carve((size_t)BQ * NQ * NQ * 4);
  float*          buf0   = (float*)carve((size_t)BQ * NQ * NQ * 4);
  float*          buf1   = (float*)carve((size_t)BQ * NQ * NQ * 4);
  float*          rvec   = (float*)carve((size_t)BQ * NQ * 4);
  float*          alphas = (float*)carve((size_t)BQ * NQ * 4);
  float*          dualIn = (float*)carve((size_t)BQ * NQ * 4);
  float*          dualOut= (float*)carve((size_t)BQ * NQ * 4);
  float*          scal   = (float*)carve((size_t)2 * BQ * 4);
  int*            flags  = (int*)carve(32 * 4);

  k_convert<<<256, 256, 0, stream>>>(gat_W, dec_W, beta_W, gatWB, decWB, betWB);
  k_encoder<<<(BQ * NQ + 255) / 256, 256, 0, stream>>>(node_input, node_emb, enc_W, enc_b, encF, encB);

  for (int layer = 0; layer < 2; ++layer) {
    k_gat_h<<<BQ * HQ * 8, 256, 0, stream>>>(encB, gatWB, gat_b, a_self, a_neigh, hB, fS, fN);
    k_gat_attn<<<BQ * 64, 128, 0, stream>>>(hB, fS, fN, node_bias, attnO);
    k_gate<<<(BQ * NQ * EQ) / 256, 256, 0, stream>>>(attnO, encF, encB, gate_w1, gate_w2, gate_b);
  }

  k_pred_softmax<<<BQ * 64, 256, 0, stream>>>(encB, decWB, dec_b, node_bias, fwp);
  k_alphas<<<(BQ * NQ + 255) / 256, 256, 0, stream>>>(encF, alpha_W, alpha_b, alphas);
  k_init<<<(BQ * NQ * NQ) / 256, 256, 0, stream>>>(buf0, dualIn, dualOut, scal, flags, out);
  k_dual<<<BQ * 64, 256, 0, stream>>>(encB, betWB, beta_b, adj, alphas, dualIn, dualOut, scal);
  k_dual_final<<<BQ, 256, 0, stream>>>(alphas, dualIn, dualOut, node_input, scal, out);

  for (int it = 0; it < ITERSQ; ++it) {
    const float* cur = (it & 1) ? buf1 : buf0;
    float* nxt = (it & 1) ? buf0 : buf1;
    k_inflow<<<BQ * 4, 256, 0, stream>>>(adj, cur, node_input, rvec, flags, it);
    k_flow_update<<<(BQ * NQ * NQ) / 256, 256, 0, stream>>>(fwp, rvec, cur, nxt, flags, it);
  }
  k_finalize<<<(BQ * NQ * NQ) / 256, 256, 0, stream>>>(buf0, buf1, flags, out);
  k_loss<<<1, 1, 0, stream>>>(out);
}
